// EnhancedODEFunction_70282844832245
// MI455X (gfx1250) — matrix operand order repaired and checked by reading, not compiled, not run on hardware
//
#include <hip/hip_runtime.h>
#include <math.h>

#define NN   50000
#define DIM  128
#define NH   4
#define CH   32
#define EDIM 16
#define NE   800000

typedef __attribute__((ext_vector_type(16))) __bf16 v16bf;
typedef __attribute__((ext_vector_type(8)))  float  v8f;

// ---------- float <-> order-preserving uint (for atomicMax on float) ----------
__device__ inline unsigned fenc(float f) {
    unsigned u = __float_as_uint(f);
    return (u & 0x80000000u) ? ~u : (u | 0x80000000u);
}
__device__ inline float fdec(unsigned u) {
    unsigned b = (u & 0x80000000u) ? (u ^ 0x80000000u) : ~u;
    return __uint_as_float(b);
}

// ---------- WMMA tile loaders: fp32 -> bf16 hi/lo split ----------
// A tile 16x32 (MxK): lane&15 = M, element i -> K = (i&7) + ((i>>3)<<4) + ((lane>>4)<<3)
__device__ inline void load_a_bf16(const float* __restrict__ base, int ld, int row0,
                                   int kc, int lane, v16bf& hi, v16bf& lo) {
    const int m = lane & 15, hf = lane >> 4;
#pragma unroll
    for (int i = 0; i < 16; ++i) {
        const int k = (i & 7) + ((i >> 3) << 4) + (hf << 3);
        float f = base[(size_t)(row0 + m) * ld + kc + k];
        __bf16 h = (__bf16)f;
        hi[i] = h;
        lo[i] = (__bf16)(f - (float)h);
    }
}
// B tile 32x16 (KxN): lane&15 = N, element i -> K = (i&7) + ((i>>3)<<4) + ((lane>>4)<<3)
__device__ inline void load_b_bf16(const float* __restrict__ W, int ld, int kc,
                                   int col0, int lane, v16bf& hi, v16bf& lo) {
    const int n = lane & 15, hf = lane >> 4;
#pragma unroll
    for (int i = 0; i < 16; ++i) {
        const int k = (i & 7) + ((i >> 3) << 4) + (hf << 3);
        float f = W[(size_t)(kc + k) * ld + col0 + n];
        __bf16 h = (__bf16)f;
        hi[i] = h;
        lo[i] = (__bf16)(f - (float)h);
    }
}
// 3-term split-precision accumulate (hi*hi + hi*lo + lo*hi) ~ fp32 accuracy
__device__ inline v8f wmma3(v16bf ah, v16bf al, v16bf bh, v16bf bl, v8f c) {
    c = __builtin_amdgcn_wmma_f32_16x16x32_bf16(false, ah, false, bh, (short)0, c, false, false);
    c = __builtin_amdgcn_wmma_f32_16x16x32_bf16(false, ah, false, bl, (short)0, c, false, false);
    c = __builtin_amdgcn_wmma_f32_16x16x32_bf16(false, al, false, bh, (short)0, c, false, false);
    return c;
}

// ---------- kernel 1: init segment-max / denom buffers ----------
__global__ void init_seg_kernel(unsigned* __restrict__ Menc, float* __restrict__ Den) {
    int i = blockIdx.x * blockDim.x + threadIdx.x;
    if (i < NN * NH) { Menc[i] = fenc(-3.0e38f); Den[i] = 0.0f; }
}

// ---------- kernel 2: Q/K/V and H0 = x@Ws+bs via WMMA ----------
// One wave per (16-row strip, matrix). A fragments (x strip, hi/lo split)
// are loaded/converted ONCE and reused across all 8 column tiles.
__global__ void node_linear_kernel(const float* __restrict__ x,
    const float* __restrict__ Wq, const float* __restrict__ bq,
    const float* __restrict__ Wk, const float* __restrict__ bk,
    const float* __restrict__ Wv, const float* __restrict__ bv,
    const float* __restrict__ Ws, const float* __restrict__ bs,
    float* __restrict__ Q, float* __restrict__ K,
    float* __restrict__ V, float* __restrict__ H0) {
    const int lane = threadIdx.x & 31;
    const int mat = blockIdx.y;
    const int row0 = blockIdx.x * 16;
    const float* W; const float* bias; float* out;
    if (mat == 0)      { W = Wq; bias = bq; out = Q; }
    else if (mat == 1) { W = Wk; bias = bk; out = K; }
    else if (mat == 2) { W = Wv; bias = bv; out = V; }
    else               { W = Ws; bias = bs; out = H0; }

    v16bf ah[4], al[4];
#pragma unroll
    for (int c = 0; c < 4; ++c)
        load_a_bf16(x, DIM, row0, c * 32, lane, ah[c], al[c]);

    const int hf = lane >> 4, n = lane & 15;
    for (int ct = 0; ct < 8; ++ct) {
        const int col0 = ct * 16;
        v8f acc = {};
#pragma unroll
        for (int c = 0; c < 4; ++c) {
            v16bf bh, bl;
            load_b_bf16(W, DIM, c * 32, col0, lane, bh, bl);
            acc = wmma3(ah[c], al[c], bh, bl, acc);
        }
        const int col = col0 + n;
        const float bv_ = bias[col];
#pragma unroll
        for (int j = 0; j < 8; ++j) {
            int row = row0 + j + hf * 8;
            out[(size_t)row * DIM + col] = acc[j] + bv_;
        }
    }
}

// ---------- kernel 3: alpha + segment max (one wave per edge, We in LDS) ----------
__global__ void edge_alpha_kernel(const int* __restrict__ ei,
                                  const float* __restrict__ eattr,
                                  const float* __restrict__ We,
                                  const float* __restrict__ Q,
                                  const float* __restrict__ K,
                                  float* __restrict__ ALPHA,
                                  unsigned* __restrict__ Menc) {
    __shared__ float sWe[EDIM * DIM];
    for (int i = threadIdx.x; i < EDIM * DIM; i += blockDim.x) sWe[i] = We[i];
    __syncthreads();

    const int lane = threadIdx.x & 31;
    const int e = blockIdx.x * 8 + (threadIdx.x >> 5);
    const int s = ei[e], d = ei[NE + e];

    float myea = (lane < EDIM) ? eattr[(size_t)e * EDIM + lane] : 0.0f;

    float part[NH];
#pragma unroll
    for (int h = 0; h < NH; ++h) {
        const int c = h * CH + lane;
        float ehc = 0.0f;
#pragma unroll
        for (int j = 0; j < EDIM; ++j)
            ehc = fmaf(__shfl(myea, j), sWe[j * DIM + c], ehc);
        float kj = K[(size_t)s * DIM + c] + ehc;
        part[h] = Q[(size_t)d * DIM + c] * kj;
    }
#pragma unroll
    for (int off = 16; off; off >>= 1)
#pragma unroll
        for (int h = 0; h < NH; ++h) part[h] += __shfl_xor(part[h], off);

    if (lane == 0) {
#pragma unroll
        for (int h = 0; h < NH; ++h) {
            float av = part[h] * 0.17677669529663687f;  // 1/sqrt(32)
            ALPHA[(size_t)e * NH + h] = av;
            atomicMax(&Menc[d * NH + h], fenc(av));
        }
    }
}

// ---------- kernel 4: ea = exp(alpha - m), segment-sum denom ----------
__global__ void edge_exp_kernel(const int* __restrict__ ei,
                                float* __restrict__ ALPHA,
                                const unsigned* __restrict__ Menc,
                                float* __restrict__ Den) {
    int t = blockIdx.x * blockDim.x + threadIdx.x;   // t in [0, NE*NH)
    int e = t >> 2, h = t & 3;
    int d = ei[NE + e];
    float m = fdec(Menc[d * NH + h]);
    float ea = __expf(ALPHA[t] - m);
    ALPHA[t] = ea;
    atomicAdd(&Den[d * NH + h], ea);
}

// ---------- kernel 5: message scatter (one wave per edge) ----------
__global__ void edge_agg_kernel(const int* __restrict__ ei,
                                const float* __restrict__ eattr,
                                const float* __restrict__ We,
                                const float* __restrict__ V,
                                const float* __restrict__ ALPHA,
                                const float* __restrict__ Den,
                                float* __restrict__ H0) {
    __shared__ float sWe[EDIM * DIM];
    for (int i = threadIdx.x; i < EDIM * DIM; i += blockDim.x) sWe[i] = We[i];
    __syncthreads();

    const int lane = threadIdx.x & 31;
    const int e = blockIdx.x * 8 + (threadIdx.x >> 5);
    const int s = ei[e], d = ei[NE + e];

    float myea = (lane < EDIM) ? eattr[(size_t)e * EDIM + lane] : 0.0f;

#pragma unroll
    for (int h = 0; h < NH; ++h) {
        const int c = h * CH + lane;
        float ehc = 0.0f;
#pragma unroll
        for (int j = 0; j < EDIM; ++j)
            ehc = fmaf(__shfl(myea, j), sWe[j * DIM + c], ehc);
        float a = ALPHA[(size_t)e * NH + h] / (Den[d * NH + h] + 1e-16f);
        float msg = (V[(size_t)s * DIM + c] + ehc) * a;
        atomicAdd(&H0[(size_t)d * DIM + c], msg);
    }
}

// ---------- kernel 6: LayerNorm1 + exact GELU ----------
__global__ void ln_gelu_kernel(const float* __restrict__ H0,
                               const float* __restrict__ g1, const float* __restrict__ b1,
                               float* __restrict__ Hg) {
    const int r = blockIdx.x, t = threadIdx.x;           // 128 threads/row
    float v = H0[(size_t)r * DIM + t];
    float s = v, s2 = v * v;
#pragma unroll
    for (int off = 16; off; off >>= 1) { s += __shfl_xor(s, off); s2 += __shfl_xor(s2, off); }
    __shared__ float rs[4], rs2[4];
    const int wid = t >> 5, lane = t & 31;
    if (lane == 0) { rs[wid] = s; rs2[wid] = s2; }
    __syncthreads();
    float sum = rs[0] + rs[1] + rs[2] + rs[3];
    float sum2 = rs2[0] + rs2[1] + rs2[2] + rs2[3];
    float mu = sum * (1.0f / DIM);
    float var = sum2 * (1.0f / DIM) - mu * mu;
    float ln = (v - mu) * rsqrtf(var + 1e-5f) * g1[t] + b1[t];
    Hg[(size_t)r * DIM + t] = 0.5f * ln * (1.0f + erff(ln * 0.70710678118654752f));
}

// ---------- kernel 7: skip = x@Wsp+bsp, gate = sigmoid([h|x]@Wg+bg), blend (WMMA) ----------
// One wave per 16-row strip; x and Hg A-fragments loaded/split once,
// reused across all 8 column tiles (skip GEMM K=128, gate GEMM K=256).
__global__ void skip_gate_kernel(const float* __restrict__ x,
                                 const float* __restrict__ Hg,
                                 const float* __restrict__ Wsp, const float* __restrict__ bsp,
                                 const float* __restrict__ Wg,  const float* __restrict__ bg,
                                 float* __restrict__ Hp) {
    const int lane = threadIdx.x & 31;
    const int row0 = blockIdx.x * 16;

    v16bf xh[4], xl[4], hh[4], hl[4];
#pragma unroll
    for (int c = 0; c < 4; ++c) {
        load_a_bf16(x,  DIM, row0, c * 32, lane, xh[c], xl[c]);
        load_a_bf16(Hg, DIM, row0, c * 32, lane, hh[c], hl[c]);
    }

    const int hf = lane >> 4, n = lane & 15;
    for (int ct = 0; ct < 8; ++ct) {
        const int col0 = ct * 16;

        v8f accS = {};
#pragma unroll
        for (int c = 0; c < 4; ++c) {
            v16bf bh, bl;
            load_b_bf16(Wsp, DIM, c * 32, col0, lane, bh, bl);
            accS = wmma3(xh[c], xl[c], bh, bl, accS);
        }
        v8f accG = {};
#pragma unroll
        for (int c = 0; c < 4; ++c) {          // rows 0..127 of Wg multiply h
            v16bf bh, bl;
            load_b_bf16(Wg, DIM, c * 32, col0, lane, bh, bl);
            accG = wmma3(hh[c], hl[c], bh, bl, accG);
        }
#pragma unroll
        for (int c = 0; c < 4; ++c) {          // rows 128..255 of Wg multiply x
            v16bf bh, bl;
            load_b_bf16(Wg, DIM, DIM + c * 32, col0, lane, bh, bl);
            accG = wmma3(xh[c], xl[c], bh, bl, accG);
        }

        const int col = col0 + n;
        const float bspv = bsp[col], bgv = bg[col];
#pragma unroll
        for (int j = 0; j < 8; ++j) {
            int row = row0 + j + hf * 8;
            float sk = accS[j] + bspv;
            float g = 1.0f / (1.0f + __expf(-(accG[j] + bgv)));
            float hg = Hg[(size_t)row * DIM + col];
            Hp[(size_t)row * DIM + col] = g * hg + (1.0f - g) * sk;
        }
    }
}

// ---------- kernel 8: LayerNorm2 -> output ----------
__global__ void ln2_kernel(const float* __restrict__ Hp,
                           const float* __restrict__ g2, const float* __restrict__ b2,
                           float* __restrict__ out) {
    const int r = blockIdx.x, t = threadIdx.x;
    float v = Hp[(size_t)r * DIM + t];
    float s = v, s2 = v * v;
#pragma unroll
    for (int off = 16; off; off >>= 1) { s += __shfl_xor(s, off); s2 += __shfl_xor(s2, off); }
    __shared__ float rs[4], rs2[4];
    const int wid = t >> 5, lane = t & 31;
    if (lane == 0) { rs[wid] = s; rs2[wid] = s2; }
    __syncthreads();
    float sum = rs[0] + rs[1] + rs[2] + rs[3];
    float sum2 = rs2[0] + rs2[1] + rs2[2] + rs2[3];
    float mu = sum * (1.0f / DIM);
    float var = sum2 * (1.0f / DIM) - mu * mu;
    out[(size_t)r * DIM + t] = (v - mu) * rsqrtf(var + 1e-5f) * g2[t] + b2[t];
}

extern "C" void kernel_launch(void* const* d_in, const int* in_sizes, int n_in,
                              void* d_out, int out_size, void* d_ws, size_t ws_size,
                              hipStream_t stream) {
    const float* x     = (const float*)d_in[0];
    const int*   ei    = (const int*)  d_in[1];
    const float* eattr = (const float*)d_in[2];
    const float* Wq = (const float*)d_in[3],  *bq = (const float*)d_in[4];
    const float* Wk = (const float*)d_in[5],  *bk = (const float*)d_in[6];
    const float* Wv = (const float*)d_in[7],  *bv = (const float*)d_in[8];
    const float* We = (const float*)d_in[9];
    const float* Ws = (const float*)d_in[10], *bs = (const float*)d_in[11];
    const float* Wsp= (const float*)d_in[12], *bsp= (const float*)d_in[13];
    const float* Wg = (const float*)d_in[14], *bg = (const float*)d_in[15];
    const float* g1 = (const float*)d_in[16], *b1 = (const float*)d_in[17];
    const float* g2 = (const float*)d_in[18], *b2 = (const float*)d_in[19];
    float* out = (float*)d_out;

    const size_t nd = (size_t)NN * DIM;
    float* Q     = (float*)d_ws;
    float* K     = Q + nd;
    float* V     = K + nd;
    float* H0    = V + nd;
    float* Hg    = H0 + nd;
    float* ALPHA = Hg + nd;                        // NE*NH floats
    float* Den   = ALPHA + (size_t)NE * NH;        // NN*NH floats
    unsigned* Menc = (unsigned*)(Den + (size_t)NN * NH);
    float* Hp = Q;                                 // Q dead after edge phase

    init_seg_kernel<<<(NN * NH + 255) / 256, 256, 0, stream>>>(Menc, Den);
    node_linear_kernel<<<dim3(NN / 16, 4), 32, 0, stream>>>(
        x, Wq, bq, Wk, bk, Wv, bv, Ws, bs, Q, K, V, H0);
    edge_alpha_kernel<<<NE / 8, 256, 0, stream>>>(ei, eattr, We, Q, K, ALPHA, Menc);
    edge_exp_kernel<<<(NE * NH) / 256, 256, 0, stream>>>(ei, ALPHA, Menc, Den);
    edge_agg_kernel<<<NE / 8, 256, 0, stream>>>(ei, eattr, We, V, ALPHA, Den, H0);
    ln_gelu_kernel<<<NN, 128, 0, stream>>>(H0, g1, b1, Hg);
    skip_gate_kernel<<<NN / 16, 32, 0, stream>>>(x, Hg, Wsp, bsp, Wg, bg, Hp);
    ln2_kernel<<<NN, 128, 0, stream>>>(Hp, g2, b2, out);
    (void)in_sizes; (void)n_in; (void)out_size; (void)ws_size;
}
